// MultilayerGRU_16690242912892
// MI455X (gfx1250) — compile-verified, matrix operand order checked by reading, not run on hardware
//
#include <hip/hip_runtime.h>
#include <hip/hip_bf16.h>

typedef __attribute__((ext_vector_type(16))) __bf16 v16bf;
typedef __attribute__((ext_vector_type(8)))  __bf16 v8bf;
typedef __attribute__((ext_vector_type(8)))  float  v8f;

#define GRU_B 32
#define GRU_S 512
#define GRU_I 256
#define GRU_H 1024
#define GRU_O 256

// ---------------------------------------------------------------------------
// Tile loader: 16x32 bf16 operand tile (A or B^T) from a row-major matrix.
// ISA layout (CDNA5 ISA, 16-bit A 16x32):
//   lane<16 : row = lane,    holds K = [k0..k0+7] (elems 0..7), [k0+16..k0+23] (elems 8..15)
//   lane>=16: row = lane-16, holds K = [k0+8..k0+15], [k0+24..k0+31]
// With W pre-transposed to [N][K], the B operand uses the identical pattern.
// ---------------------------------------------------------------------------
static __device__ inline v16bf load_tile16x32(const __bf16* __restrict__ p, int ld,
                                              int row0, int k0, int lane) {
  const int r  = row0 + (lane & 15);
  const int kb = k0 + ((lane >> 4) << 3);
  const __bf16* q = p + (size_t)r * ld + kb;
  v8bf lo = *(const v8bf*)(q);
  v8bf hi = *(const v8bf*)(q + 16);
  v16bf out;
#pragma unroll
  for (int i = 0; i < 8; ++i) { out[i] = lo[i]; out[i + 8] = hi[i]; }
  return out;
}

static __device__ inline v8f wmma_bf16(v16bf a, v16bf b, v8f c) {
  return __builtin_amdgcn_wmma_f32_16x16x32_bf16(false, a, false, b, (short)0, c,
                                                 false, false);
}

// C += A[arow..arow+16][0..K) * Bt[brow..brow+16][0..K)^T
static __device__ inline v8f gemm_tile(const __bf16* __restrict__ A, int lda, int arow,
                                       const __bf16* __restrict__ Bt, int ldb, int brow,
                                       int K, v8f c, int lane) {
  for (int k = 0; k < K; k += 32) {
    v16bf a = load_tile16x32(A, lda, arow, k, lane);
    v16bf b = load_tile16x32(Bt, ldb, brow, k, lane);
    c = wmma_bf16(a, b, c);
  }
  return c;
}

static __device__ inline float sigmoidf_(float x) {
  return 1.0f / (1.0f + __expf(-x));
}

// Grid-wide split barrier through L2 (counter + generation).
static __device__ inline void grid_barrier(unsigned* ctr, unsigned* gen, unsigned nwg) {
  __syncthreads();
  if (threadIdx.x == 0) {
    __threadfence();
    unsigned g = __hip_atomic_load(gen, __ATOMIC_RELAXED, __HIP_MEMORY_SCOPE_AGENT);
    unsigned prev = __hip_atomic_fetch_add(ctr, 1u, __ATOMIC_ACQ_REL, __HIP_MEMORY_SCOPE_AGENT);
    if (prev == nwg - 1u) {
      __hip_atomic_store(ctr, 0u, __ATOMIC_RELAXED, __HIP_MEMORY_SCOPE_AGENT);
      __hip_atomic_store(gen, g + 1u, __ATOMIC_RELEASE, __HIP_MEMORY_SCOPE_AGENT);
    } else {
      while (__hip_atomic_load(gen, __ATOMIC_ACQUIRE, __HIP_MEMORY_SCOPE_AGENT) == g)
        __builtin_amdgcn_s_sleep(1);
    }
    __threadfence();
  }
  __syncthreads();
}

// ---------------------------------------------------------------------------
// Prep kernels
// ---------------------------------------------------------------------------
__global__ void k_init(__bf16* h0, __bf16* h1, unsigned* sync) {
  int i = blockIdx.x * blockDim.x + threadIdx.x;
  if (i < GRU_B * GRU_H) { h0[i] = (__bf16)0.0f; h1[i] = (__bf16)0.0f; }
  if (i == 0) { sync[0] = 0u; sync[1] = 0u; }
}

__global__ void k_cvt(const float* __restrict__ src, __bf16* __restrict__ dst, int n) {
  int stride = gridDim.x * blockDim.x;
  for (int i = blockIdx.x * blockDim.x + threadIdx.x; i < n; i += stride)
    dst[i] = (__bf16)src[i];
}

// dst[n*K + k] = (bf16) src[k*N + n]   (store W^T for contiguous B-operand loads)
__global__ void k_cvt_t(const float* __restrict__ src, __bf16* __restrict__ dst,
                        int K, int N) {
  int total = K * N;
  int stride = gridDim.x * blockDim.x;
  for (int i = blockIdx.x * blockDim.x + threadIdx.x; i < total; i += stride) {
    int n = i / K, k = i - n * K;
    dst[i] = (__bf16)src[(size_t)k * N + n];
  }
}

// ---------------------------------------------------------------------------
// Batched input projections for layer 0:
//   Xp[gate][s][b][h] = x[b][s][:] @ Wx{gate}0 + b{gate}0
// A rows r = b*512 + s (row-major x), N = 3*H via stacked transposed weights.
// ---------------------------------------------------------------------------
__global__ void k_xproj(const __bf16* __restrict__ xb,   // [B*S][I]
                        const __bf16* __restrict__ WX0t, // [3*H][I]
                        const float* __restrict__ bz0, const float* __restrict__ br0,
                        const float* __restrict__ bg0,
                        float* __restrict__ Xp) {        // [3][S][B][H]
  const int lane = threadIdx.x & 31;
  const int wave = threadIdx.x >> 5;
  const int mt = blockIdx.x;              // 0..1023
  const int nt = blockIdx.y * 4 + wave;   // 0..191
  const int gate = nt >> 6;
  v8f c = {};
  c = gemm_tile(xb, GRU_I, mt * 16, WX0t, GRU_I, nt * 16, GRU_I, c, lane);
  const float* bias = (gate == 0) ? bz0 : (gate == 1 ? br0 : bg0);
  const int hcol = (nt & 63) * 16 + (lane & 15);
  const float bv = bias[hcol];
#pragma unroll
  for (int i = 0; i < 8; ++i) {
    int r = mt * 16 + i + ((lane >> 4) << 3);
    int b = r >> 9, s = r & 511;
    Xp[(((size_t)gate * GRU_S + s) * GRU_B + b) * GRU_H + hcol] = c[i] + bv;
  }
}

// ---------------------------------------------------------------------------
// Persistent recurrent kernel: 64 WGs x 4 waves. WG owns 16 output columns.
// Wave roles: (mtile = wave&1 -> batch rows 0-15/16-31), (grole = wave>>1: z / r).
// Per step: A) z0,r0  B) g0+combine  C) z1,r1 (K=2048)  D) g1+combine.
// 4 grid barriers per step.
// ---------------------------------------------------------------------------
__global__ void __launch_bounds__(128, 1)
k_gru(const __bf16* __restrict__ Wh0t,  // [3][H][H]  (z,r,g) transposed
      const __bf16* __restrict__ WX1t,  // [3][H][H]
      const __bf16* __restrict__ Wh1t,  // [3][H][H]
      const float* __restrict__ Xp,     // [3][S][B][H]
      const float* __restrict__ bz1, const float* __restrict__ br1,
      const float* __restrict__ bg1,
      __bf16* __restrict__ h0, __bf16* __restrict__ h1, __bf16* __restrict__ rh,
      __bf16* __restrict__ h1a,         // [S][B][H]
      float* __restrict__ hid_out,      // [B][2][H]
      unsigned* __restrict__ ctr, unsigned* __restrict__ gen, int nwg) {
  const int lane  = threadIdx.x & 31;
  const int wave  = threadIdx.x >> 5;
  const int mtile = wave & 1;
  const int grole = wave >> 1;          // 0 = z-role, 1 = r-role
  const int col0  = blockIdx.x * 16;
  const int ncol  = col0 + (lane & 15);
  const int mrow0 = mtile * 16;
  const size_t HH = (size_t)GRU_H * GRU_H;

  float zk0[8], zk1[8];

  for (int t = 0; t < GRU_S; ++t) {
    // ---- Layer 0, phase A: z0 (grole 0) / r0 (grole 1)
    {
      v8f c = {};
      c = gemm_tile(h0, GRU_H, mrow0, Wh0t + (size_t)grole * HH, GRU_H, col0,
                    GRU_H, c, lane);
      const float* xp = Xp + ((size_t)grole * GRU_S + t) * (GRU_B * GRU_H);
#pragma unroll
      for (int i = 0; i < 8; ++i) {
        int M = mrow0 + i + ((lane >> 4) << 3);
        float s = sigmoidf_(c[i] + xp[M * GRU_H + ncol]);
        if (grole == 0) zk0[i] = s;
        else rh[M * GRU_H + ncol] = (__bf16)(s * (float)h0[M * GRU_H + ncol]);
      }
    }
    grid_barrier(ctr, gen, nwg);
    // ---- Layer 0, phase B: g0 = tanh(Xpg + (r*h)@Whg0), h0' = z*h + (1-z)*g
    if (grole == 0) {
      v8f c = {};
      c = gemm_tile(rh, GRU_H, mrow0, Wh0t + 2 * HH, GRU_H, col0, GRU_H, c, lane);
      const float* xp = Xp + ((size_t)2 * GRU_S + t) * (GRU_B * GRU_H);
#pragma unroll
      for (int i = 0; i < 8; ++i) {
        int M = mrow0 + i + ((lane >> 4) << 3);
        float g  = tanhf(c[i] + xp[M * GRU_H + ncol]);
        float hv = (float)h0[M * GRU_H + ncol];
        float hn = zk0[i] * hv + (1.0f - zk0[i]) * g;
        h0[M * GRU_H + ncol] = (__bf16)hn;
        if (t == GRU_S - 1) hid_out[M * 2 * GRU_H + ncol] = hn;
      }
    }
    grid_barrier(ctr, gen, nwg);
    // ---- Layer 1, phase C: z1/r1 = sig(h0n@Wx?1 + h1@Wh?1 + b?1), K=2048
    {
      v8f c = {};
      c = gemm_tile(h0, GRU_H, mrow0, WX1t + (size_t)grole * HH, GRU_H, col0,
                    GRU_H, c, lane);
      c = gemm_tile(h1, GRU_H, mrow0, Wh1t + (size_t)grole * HH, GRU_H, col0,
                    GRU_H, c, lane);
      const float bv = (grole ? br1 : bz1)[ncol];
#pragma unroll
      for (int i = 0; i < 8; ++i) {
        int M = mrow0 + i + ((lane >> 4) << 3);
        float s = sigmoidf_(c[i] + bv);
        if (grole == 0) zk1[i] = s;
        else rh[M * GRU_H + ncol] = (__bf16)(s * (float)h1[M * GRU_H + ncol]);
      }
    }
    grid_barrier(ctr, gen, nwg);
    // ---- Layer 1, phase D: g1 = tanh(h0n@Wxg1 + (r1*h1)@Whg1 + bg1), combine
    if (grole == 0) {
      v8f c = {};
      c = gemm_tile(h0, GRU_H, mrow0, WX1t + 2 * HH, GRU_H, col0, GRU_H, c, lane);
      c = gemm_tile(rh, GRU_H, mrow0, Wh1t + 2 * HH, GRU_H, col0, GRU_H, c, lane);
      const float bv = bg1[ncol];
#pragma unroll
      for (int i = 0; i < 8; ++i) {
        int M = mrow0 + i + ((lane >> 4) << 3);
        float g  = tanhf(c[i] + bv);
        float hv = (float)h1[M * GRU_H + ncol];
        float hn = zk1[i] * hv + (1.0f - zk1[i]) * g;
        __bf16 hb = (__bf16)hn;
        h1[M * GRU_H + ncol] = hb;
        h1a[((size_t)t * GRU_B + M) * GRU_H + ncol] = hb;
        if (t == GRU_S - 1) hid_out[M * 2 * GRU_H + GRU_H + ncol] = hn;
      }
    }
    grid_barrier(ctr, gen, nwg);
  }
}

// ---------------------------------------------------------------------------
// Batched output projection: y[b][t][:] = h1a[t][b][:] @ Why + bhy
// A rows r = t*32 + b.
// ---------------------------------------------------------------------------
__global__ void k_yout(const __bf16* __restrict__ h1a,  // [S*B][H]
                       const __bf16* __restrict__ Whyt, // [O][H]
                       const float* __restrict__ bhy,
                       float* __restrict__ y) {         // [B][S][O]
  const int lane = threadIdx.x & 31;
  const int wave = threadIdx.x >> 5;
  const int mt = blockIdx.x;             // 0..1023
  const int nt = blockIdx.y * 4 + wave;  // 0..15
  v8f c = {};
  c = gemm_tile(h1a, GRU_H, mt * 16, Whyt, GRU_H, nt * 16, GRU_H, c, lane);
  const int ncol = nt * 16 + (lane & 15);
  const float bv = bhy[ncol];
#pragma unroll
  for (int i = 0; i < 8; ++i) {
    int r = mt * 16 + i + ((lane >> 4) << 3);
    int t = r >> 5, b = r & 31;
    y[((size_t)b * GRU_S + t) * GRU_O + ncol] = c[i] + bv;
  }
}

// ---------------------------------------------------------------------------
extern "C" void kernel_launch(void* const* d_in, const int* in_sizes, int n_in,
                              void* d_out, int out_size, void* d_ws, size_t ws_size,
                              hipStream_t stream) {
  (void)in_sizes; (void)n_in; (void)out_size; (void)ws_size;
  const float* x    = (const float*)d_in[0];
  const float* Wxz0 = (const float*)d_in[1];
  const float* Whz0 = (const float*)d_in[2];
  const float* bz0  = (const float*)d_in[3];
  const float* Wxr0 = (const float*)d_in[4];
  const float* Whr0 = (const float*)d_in[5];
  const float* br0  = (const float*)d_in[6];
  const float* Wxg0 = (const float*)d_in[7];
  const float* Whg0 = (const float*)d_in[8];
  const float* bg0  = (const float*)d_in[9];
  const float* Wxz1 = (const float*)d_in[10];
  const float* Whz1 = (const float*)d_in[11];
  const float* bz1  = (const float*)d_in[12];
  const float* Wxr1 = (const float*)d_in[13];
  const float* Whr1 = (const float*)d_in[14];
  const float* br1  = (const float*)d_in[15];
  const float* Wxg1 = (const float*)d_in[16];
  const float* Whg1 = (const float*)d_in[17];
  const float* bg1  = (const float*)d_in[18];
  const float* Why  = (const float*)d_in[19];
  const float* bhy  = (const float*)d_in[20];

  char* base = (char*)d_ws;
  size_t off = 0;
  auto take = [&](size_t bytes) -> char* {
    char* p = base + off;
    off += (bytes + 255) & ~(size_t)255;
    return p;
  };

  __bf16* xb   = (__bf16*)take((size_t)GRU_B * GRU_S * GRU_I * 2);
  __bf16* WX0t = (__bf16*)take((size_t)3 * GRU_H * GRU_I * 2);
  __bf16* Wh0t = (__bf16*)take((size_t)3 * GRU_H * GRU_H * 2);
  __bf16* WX1t = (__bf16*)take((size_t)3 * GRU_H * GRU_H * 2);
  __bf16* Wh1t = (__bf16*)take((size_t)3 * GRU_H * GRU_H * 2);
  __bf16* Whyt = (__bf16*)take((size_t)GRU_O * GRU_H * 2);
  float*  Xp   = (float*) take((size_t)3 * GRU_S * GRU_B * GRU_H * 4);
  __bf16* h0   = (__bf16*)take((size_t)GRU_B * GRU_H * 2);
  __bf16* h1   = (__bf16*)take((size_t)GRU_B * GRU_H * 2);
  __bf16* rh   = (__bf16*)take((size_t)GRU_B * GRU_H * 2);
  __bf16* h1a  = (__bf16*)take((size_t)GRU_S * GRU_B * GRU_H * 2);
  unsigned* sync = (unsigned*)take(256);

  float* y_out   = (float*)d_out;
  float* hid_out = (float*)d_out + (size_t)GRU_B * GRU_S * GRU_O;

  const int HH = GRU_H * GRU_H;
  const int HI = GRU_H * GRU_I;

  k_init<<<dim3((GRU_B * GRU_H + 255) / 256), dim3(256), 0, stream>>>(h0, h1, sync);
  k_cvt<<<dim3(512), dim3(256), 0, stream>>>(x, xb, GRU_B * GRU_S * GRU_I);

  k_cvt_t<<<dim3(256), dim3(256), 0, stream>>>(Wxz0, WX0t,          GRU_I, GRU_H);
  k_cvt_t<<<dim3(256), dim3(256), 0, stream>>>(Wxr0, WX0t + HI,     GRU_I, GRU_H);
  k_cvt_t<<<dim3(256), dim3(256), 0, stream>>>(Wxg0, WX0t + 2 * HI, GRU_I, GRU_H);
  k_cvt_t<<<dim3(1024), dim3(256), 0, stream>>>(Whz0, Wh0t,          GRU_H, GRU_H);
  k_cvt_t<<<dim3(1024), dim3(256), 0, stream>>>(Whr0, Wh0t + HH,     GRU_H, GRU_H);
  k_cvt_t<<<dim3(1024), dim3(256), 0, stream>>>(Whg0, Wh0t + 2 * HH, GRU_H, GRU_H);
  k_cvt_t<<<dim3(1024), dim3(256), 0, stream>>>(Wxz1, WX1t,          GRU_H, GRU_H);
  k_cvt_t<<<dim3(1024), dim3(256), 0, stream>>>(Wxr1, WX1t + HH,     GRU_H, GRU_H);
  k_cvt_t<<<dim3(1024), dim3(256), 0, stream>>>(Wxg1, WX1t + 2 * HH, GRU_H, GRU_H);
  k_cvt_t<<<dim3(1024), dim3(256), 0, stream>>>(Whz1, Wh1t,          GRU_H, GRU_H);
  k_cvt_t<<<dim3(1024), dim3(256), 0, stream>>>(Whr1, Wh1t + HH,     GRU_H, GRU_H);
  k_cvt_t<<<dim3(1024), dim3(256), 0, stream>>>(Whg1, Wh1t + 2 * HH, GRU_H, GRU_H);
  k_cvt_t<<<dim3(256), dim3(256), 0, stream>>>(Why, Whyt, GRU_H, GRU_O);

  k_xproj<<<dim3(1024, 48), dim3(128), 0, stream>>>(xb, WX0t, bz0, br0, bg0, Xp);
  k_gru<<<dim3(64), dim3(128), 0, stream>>>(Wh0t, WX1t, Wh1t, Xp, bz1, br1, bg1,
                                            h0, h1, rh, h1a, hid_out,
                                            sync, sync + 1, 64);
  k_yout<<<dim3(1024, 4), dim3(128), 0, stream>>>(h1a, Whyt, bhy, y_out);
}
